// SymGatedGCNMambaOnlyModel_39256001085692
// MI455X (gfx1250) — matrix operand order repaired and checked by reading, not compiled, not run on hardware
//
#include <hip/hip_runtime.h>
#include <hip/hip_bf16.h>

typedef __attribute__((ext_vector_type(16))) _Float16 v16h;
typedef __attribute__((ext_vector_type(8)))  _Float16 v8h;
typedef __attribute__((ext_vector_type(4)))  _Float16 v4h;
typedef __attribute__((ext_vector_type(8)))  float    v8f;
typedef __attribute__((ext_vector_type(4)))  float    v4f;

static __host__ int cdiv(int a, int b) { return (a + b - 1) / b; }

// ---------------------------------------------------------------------------
// Generic 64-wide GEMM with WMMA:  Y(M,64) = act(X(M,K) @ W(K,64) + bias)
// K must be a multiple of 32 (here: 64 / 128 / 192 / 256).
// Block = 128 threads = 4 waves; each wave owns a 16-row tile, all 64 cols.
// W is staged to LDS pre-swizzled into WMMA B-fragment order so each lane
// reads its 16-half fragment as one contiguous 32B LDS access.
// A_F16=0: X is fp32, A fragments from four b128 loads + packed converts.
// A_F16=1: X is f16,  A fragments from two b128 loads, no conversion.
// ---------------------------------------------------------------------------
template <int A_F16>
__global__ __launch_bounds__(128)
void gemm64_wmma_kernel(const void* __restrict__ Xv, const float* __restrict__ W,
                        const float* __restrict__ bias, float* __restrict__ Y,
                        int M, int K, int act) {
  __shared__ _Float16 sW[256 * 64];   // fragment-swizzled
  __shared__ float sBias[64];
  const int tid = threadIdx.x;
  // Stage W[k][n] -> sW[((blk*4 + nt)*32 + lane)*16 + h]
  //   blk = k>>5, hi = ((k&31)>>3)&1, h = (k&7) | (((k&31)>>4)<<3), nt = n>>4, lane = hi*16 + (n&15)
  for (int i = tid; i < K * 64; i += 128) {
    int k = i >> 6, n = i & 63;
    int blk = k >> 5, kl = k & 31;
    int lane_w = (((kl >> 3) & 1) << 4) | (n & 15);
    int idx = (((blk * 4 + (n >> 4)) * 32 + lane_w) << 4) | ((kl & 7) | ((kl >> 4) << 3));
    sW[idx] = (_Float16)W[i];
  }
  if (tid < 64) sBias[tid] = bias[tid];
  __syncthreads();

  const int wave = tid >> 5;
  const int lane = tid & 31;
  const int m  = lane & 15;   // row within tile (A) / column (B, D)
  const int hi = lane >> 4;   // high-K half selector
  int rowA = blockIdx.x * 64 + wave * 16 + m;
  if (rowA >= M) rowA = M - 1;  // clamp: out-of-range D rows are never stored
  const float*    xrf = (const float*)Xv + (size_t)rowA * K;
  const _Float16* xrh = (const _Float16*)Xv + (size_t)rowA * K;
  const v16h* bW = (const v16h*)sW;

  v8f acc[4];
#pragma unroll
  for (int nt = 0; nt < 4; ++nt) {
    float bv = sBias[nt * 16 + m];
#pragma unroll
    for (int r = 0; r < 8; ++r) acc[nt][r] = bv;
  }

  for (int k0 = 0, blk = 0; k0 < K; k0 += 32, ++blk) {
    // A 16x32 f16 fragment: halves 0..7 = K[k0+hi*8 .. +7], 8..15 = K[k0+16+hi*8 .. +7]
    v16h a;
    if (A_F16) {
      v8h g0 = *(const v8h*)(xrh + k0 + hi * 8);
      v8h g1 = *(const v8h*)(xrh + k0 + 16 + hi * 8);
#pragma unroll
      for (int j = 0; j < 8; ++j) { a[j] = g0[j]; a[8 + j] = g1[j]; }
    } else {
      const v4f* p0 = (const v4f*)(xrf + k0 + hi * 8);
      const v4f* p1 = (const v4f*)(xrf + k0 + 16 + hi * 8);
      v4f f0 = p0[0], f1 = p0[1], f2 = p1[0], f3 = p1[1];
#pragma unroll
      for (int j = 0; j < 4; ++j) {
        a[j]      = (_Float16)f0[j];
        a[4 + j]  = (_Float16)f1[j];
        a[8 + j]  = (_Float16)f2[j];
        a[12 + j] = (_Float16)f3[j];
      }
    }
    const v16h* bp = bW + (size_t)(blk * 4) * 32 + lane;
#pragma unroll
    for (int nt = 0; nt < 4; ++nt) {
      v16h b = bp[nt * 32];
      acc[nt] = __builtin_amdgcn_wmma_f32_16x16x32_f16(
          false, a, false, b, (short)0, acc[nt], false, false);
    }
  }

#pragma unroll
  for (int nt = 0; nt < 4; ++nt) {
#pragma unroll
    for (int r = 0; r < 8; ++r) {
      // D layout: VGPR r -> row r (lanes<16) / r+8 (lanes>=16); col = lane&15
      int orow = blockIdx.x * 64 + wave * 16 + r + (hi << 3);
      if (orow < M) {
        float v = acc[nt][r];
        if (act) v = fmaxf(v, 0.0f);
        Y[(size_t)orow * 64 + nt * 16 + m] = v;
      }
    }
  }
}

// ---------------------------------------------------------------------------
// Node encoder: h = relu(x @ w1n + b1n) @ w2n + b2n   (x: N x 2, HI=128, H=64)
// ---------------------------------------------------------------------------
__global__ __launch_bounds__(128)
void node_mlp_kernel(const float* __restrict__ x, const float* __restrict__ w1n,
                     const float* __restrict__ b1n, const float* __restrict__ w2n,
                     const float* __restrict__ b2n, float* __restrict__ h, int N) {
  __shared__ float s[128];
  int n = blockIdx.x;
  if (n >= N) return;
  int t = threadIdx.x;
  float x0 = x[(size_t)n * 2 + 0], x1 = x[(size_t)n * 2 + 1];
  s[t] = fmaxf(x0 * w1n[t] + x1 * w1n[128 + t] + b1n[t], 0.0f);
  __syncthreads();
  if (t < 64) {
    float acc = b2n[t];
    for (int k = 0; k < 128; ++k) acc += s[k] * w2n[k * 64 + t];
    h[(size_t)n * 64 + t] = acc;
  }
}

// ---------------------------------------------------------------------------
// Mamba kernels (dims tiny: DI=8, DS=16, DTR=1 -> scalar VALU)
// ---------------------------------------------------------------------------
__global__ void mamba_pre_kernel(const float* __restrict__ seq, const float* __restrict__ mnw,
                                 const float* __restrict__ win, float* __restrict__ xi_pre,
                                 float* __restrict__ zb, int total) {
  int i = blockIdx.x * 256 + threadIdx.x;
  if (i >= total) return;
  v4f u = ((const v4f*)seq)[i];
  float ms = u[0]*u[0] + u[1]*u[1] + u[2]*u[2] + u[3]*u[3];
  float r = rsqrtf(ms * 0.25f + 1e-5f);
  float un[4];
#pragma unroll
  for (int c = 0; c < 4; ++c) un[c] = u[c] * r * mnw[c];
  v4f o0, o1, o2, o3;
#pragma unroll
  for (int j = 0; j < 16; ++j) {
    float s = 0.f;
#pragma unroll
    for (int c = 0; c < 4; ++c) s += un[c] * win[c * 16 + j];
    if (j < 4) o0[j] = s;
    else if (j < 8) o1[j - 4] = s;
    else if (j < 12) o2[j - 8] = s;
    else o3[j - 12] = s;
  }
  ((v4f*)xi_pre)[i * 2 + 0] = o0;
  ((v4f*)xi_pre)[i * 2 + 1] = o1;
  ((v4f*)zb)[i * 2 + 0] = o2;
  ((v4f*)zb)[i * 2 + 1] = o3;
}

__global__ void mamba_conv_kernel(const float* __restrict__ xi_pre, const float* __restrict__ wconv,
                                  const float* __restrict__ bconv, const float* __restrict__ wx,
                                  const float* __restrict__ wdt, const float* __restrict__ bdt,
                                  float* __restrict__ xic, float* __restrict__ delt,
                                  float* __restrict__ Bm, float* __restrict__ Cm,
                                  int L, int total) {
  int i = blockIdx.x * 256 + threadIdx.x;
  if (i >= total) return;
  int l = i % L;
  float xc[8];
#pragma unroll
  for (int d = 0; d < 8; ++d) {
    float acc = bconv[d];
#pragma unroll
    for (int k = 0; k < 4; ++k) {
      int li = l - 3 + k;
      if (li >= 0) acc += xi_pre[(size_t)(i - l + li) * 8 + d] * wconv[d * 4 + k];
    }
    float s = acc / (1.f + expf(-acc));  // silu
    xc[d] = s;
    xic[(size_t)i * 8 + d] = s;
  }
  float dt = 0.f;
#pragma unroll
  for (int d = 0; d < 8; ++d) dt += xc[d] * wx[d * 33 + 0];
#pragma unroll
  for (int d = 0; d < 8; ++d) {
    float a = dt * wdt[d] + bdt[d];
    delt[(size_t)i * 8 + d] = (a > 20.f) ? a : log1pf(expf(a));  // softplus
  }
#pragma unroll
  for (int s = 0; s < 16; ++s) {
    float b = 0.f, c = 0.f;
#pragma unroll
    for (int d = 0; d < 8; ++d) {
      b += xc[d] * wx[d * 33 + 1 + s];
      c += xc[d] * wx[d * 33 + 17 + s];
    }
    Bm[(size_t)i * 16 + s] = b;
    Cm[(size_t)i * 16 + s] = c;
  }
}

__global__ void mamba_scan_kernel(const float* __restrict__ xic, const float* __restrict__ delt,
                                  const float* __restrict__ Bm, const float* __restrict__ Cm,
                                  const float* __restrict__ alog, const float* __restrict__ md,
                                  float* __restrict__ ytmp, int L, int total) {
  int t = blockIdx.x * 256 + threadIdx.x;
  if (t >= total) return;
  int n = t >> 3, d = t & 7;
  float A[16], hs[16];
#pragma unroll
  for (int s = 0; s < 16; ++s) { A[s] = -expf(alog[d * 16 + s]); hs[s] = 0.f; }
  float dd = md[d];
  const v4f* B4 = (const v4f*)Bm;
  const v4f* C4 = (const v4f*)Cm;
  for (int l = 0; l < L; ++l) {
    size_t idx = (size_t)n * L + l;
    float xv = xic[idx * 8 + d];
    float dl = delt[idx * 8 + d];
    float y = 0.f;
#pragma unroll
    for (int q = 0; q < 4; ++q) {
      v4f bq = B4[idx * 4 + q];
      v4f cq = C4[idx * 4 + q];
#pragma unroll
      for (int j = 0; j < 4; ++j) {
        int s = q * 4 + j;
        hs[s] = expf(dl * A[s]) * hs[s] + dl * xv * bq[j];
        y += hs[s] * cq[j];
      }
    }
    ytmp[idx * 8 + d] = y + dd * xv;
  }
}

__global__ void mamba_out_kernel(const float* __restrict__ ytmp, const float* __restrict__ zb,
                                 const float* __restrict__ wout, float* __restrict__ seq, int total) {
  int i = blockIdx.x * 256 + threadIdx.x;
  if (i >= total) return;
  float g[8];
#pragma unroll
  for (int d = 0; d < 8; ++d) {
    float zd = zb[(size_t)i * 8 + d];
    g[d] = ytmp[(size_t)i * 8 + d] * (zd / (1.f + expf(-zd)));
  }
  v4f o = ((const v4f*)seq)[i];
#pragma unroll
  for (int c = 0; c < 4; ++c) {
    float acc = 0.f;
#pragma unroll
    for (int d = 0; d < 8; ++d) acc += g[d] * wout[d * 4 + c];
    o[c] += acc;
  }
  ((v4f*)seq)[i] = o;
}

// x2 = seq[n, read_length[n]-1, :4] @ wbase + bbase
__global__ void last_token_kernel(const float* __restrict__ seq, const int* __restrict__ rl,
                                  const float* __restrict__ wbase, const float* __restrict__ bbase,
                                  float* __restrict__ x2, int N, int L) {
  int i = blockIdx.x * 256 + threadIdx.x;
  if (i >= N * 64) return;
  int n = i >> 6, j = i & 63;
  int l = rl[n] - 1;
  v4f s = *(const v4f*)&seq[((size_t)n * L + l) * 4];
  float acc = bbase[j];
#pragma unroll
  for (int c = 0; c < 4; ++c) acc += s[c] * wbase[c * 64 + j];
  x2[i] = acc;
}

// ---------------------------------------------------------------------------
// Gathers / elementwise
// ---------------------------------------------------------------------------
__global__ void concat2_kernel(const float* __restrict__ a, const float* __restrict__ b,
                               float* __restrict__ out, int N) {
  int i = blockIdx.x * 256 + threadIdx.x;   // over N*32 float4s
  if (i >= N * 32) return;
  int n = i >> 5, q = i & 31;
  const float* basep = (q < 16) ? a : b;
  ((v4f*)out)[i] = ((const v4f*)(basep + (size_t)n * 64))[q & 15];
}

// data row (f16) = [h[src] | h[dst] | x2[src] | x2[dst]]  (256 halves -> 64 quads)
__global__ void gather_edge_kernel(const float* __restrict__ h, const float* __restrict__ x2,
                                   const int* __restrict__ src, const int* __restrict__ dst,
                                   _Float16* __restrict__ data, long long total4) {
  long long i = (long long)blockIdx.x * 256 + threadIdx.x;
  if (i >= total4) return;
  int e = (int)(i >> 6), q = (int)(i & 63);
  int seg = q >> 4;
  int node = (seg == 0 || seg == 2) ? src[e] : dst[e];
  const float* basep = (seg < 2) ? h : x2;
  v4f v = ((const v4f*)(basep + (size_t)node * 64))[q & 15];
  v4h o;
#pragma unroll
  for (int j = 0; j < 4; ++j) o[j] = (_Float16)v[j];
  ((v4h*)data)[i] = o;
}

// pdata row (f16) = [hc[src] | hc[dst] | e]  (192 halves -> 48 quads)
__global__ void gather_pdata_kernel(const float* __restrict__ hc, const float* __restrict__ ebuf,
                                    const int* __restrict__ src, const int* __restrict__ dst,
                                    _Float16* __restrict__ pdata, long long total4) {
  long long i = (long long)blockIdx.x * 256 + threadIdx.x;
  if (i >= total4) return;
  int e = (int)(i / 48), q = (int)(i % 48);
  int seg = q >> 4;
  v4f v;
  if (seg == 0)      v = ((const v4f*)(hc + (size_t)src[e] * 64))[q & 15];
  else if (seg == 1) v = ((const v4f*)(hc + (size_t)dst[e] * 64))[q & 15];
  else               v = ((const v4f*)(ebuf + (size_t)e * 64))[q & 15];
  v4h o;
#pragma unroll
  for (int j = 0; j < 4; ++j) o[j] = (_Float16)v[j];
  ((v4h*)pdata)[i] = o;
}

__global__ void zero_kernel(float* __restrict__ p, long long n4) {
  long long i = (long long)blockIdx.x * 256 + threadIdx.x;
  if (i < n4) ((v4f*)p)[i] = (v4f){0.f, 0.f, 0.f, 0.f};
}

// ---------------------------------------------------------------------------
// GNN layer kernels
// ---------------------------------------------------------------------------
__global__ __launch_bounds__(256)
void gnn_edge_kernel(float* __restrict__ eh, const float* __restrict__ Q2,
                     const float* __restrict__ Q3, const float* __restrict__ P2,
                     const float* __restrict__ P3, const int* __restrict__ src,
                     const int* __restrict__ dst, float* __restrict__ nf,
                     float* __restrict__ df, float* __restrict__ nb,
                     float* __restrict__ db, float* __restrict__ stats, int E) {
  __shared__ float s1[64], s2[64];
  int t = threadIdx.x;
  if (t < 64) { s1[t] = 0.f; s2[t] = 0.f; }
  __syncthreads();
  int c = t & 63;
  int e = blockIdx.x * 4 + (t >> 6);
  if (e < E) {
    int s = src[e], d = dst[e];
    float v = eh[(size_t)e * 64 + c] + Q2[(size_t)s * 64 + c] + Q3[(size_t)d * 64 + c];
    eh[(size_t)e * 64 + c] = v;
    float sig = 1.0f / (1.0f + expf(-v));
    atomicAdd(&nf[(size_t)d * 64 + c], sig * P2[(size_t)s * 64 + c]);
    atomicAdd(&df[(size_t)d * 64 + c], sig);
    atomicAdd(&nb[(size_t)s * 64 + c], sig * P3[(size_t)d * 64 + c]);
    atomicAdd(&db[(size_t)s * 64 + c], sig);
    atomicAdd(&s1[c], v);
    atomicAdd(&s2[c], v * v);
  }
  __syncthreads();
  if (t < 64) { atomicAdd(&stats[t], s1[t]); atomicAdd(&stats[64 + t], s2[t]); }
}

__global__ void bn_finalize_kernel(float* __restrict__ stats, const float* __restrict__ g,
                                   const float* __restrict__ b, float invcnt) {
  int c = threadIdx.x;
  if (c < 64) {
    float mean = stats[c] * invcnt;
    float var = stats[64 + c] * invcnt - mean * mean;
    float sc = g[c] * rsqrtf(var + 1e-5f);
    stats[128 + c] = sc;
    stats[192 + c] = b[c] - mean * sc;
  }
}

__global__ __launch_bounds__(256)
void node_hagg_kernel(const float* __restrict__ P1, const float* __restrict__ nf,
                      const float* __restrict__ df, const float* __restrict__ nb,
                      const float* __restrict__ db, float* __restrict__ hagg,
                      float* __restrict__ stats, int total) {
  __shared__ float s1[64], s2[64];
  int t = threadIdx.x;
  if (t < 64) { s1[t] = 0.f; s2[t] = 0.f; }
  __syncthreads();
  int i = blockIdx.x * 256 + t;
  int c = t & 63;
  if (i < total) {
    float v = P1[i] + nf[i] / (df[i] + 1e-6f) + nb[i] / (db[i] + 1e-6f);
    hagg[i] = v;
    atomicAdd(&s1[c], v);
    atomicAdd(&s2[c], v * v);
  }
  __syncthreads();
  if (t < 64) { atomicAdd(&stats[t], s1[t]); atomicAdd(&stats[64 + t], s2[t]); }
}

// dst += relu(val * scale + shift)   (scale/shift in stats[128..255]); total multiple of 64
__global__ void bn_residual_kernel(float* __restrict__ dstbuf, const float* __restrict__ val,
                                   const float* __restrict__ stats, long long total4) {
  long long i = (long long)blockIdx.x * 256 + threadIdx.x;
  if (i >= total4) return;
  int c = (int)((i * 4) & 63);
  v4f v  = ((const v4f*)val)[i];
  v4f sc = *(const v4f*)&stats[128 + c];
  v4f sh = *(const v4f*)&stats[192 + c];
  v4f d  = ((v4f*)dstbuf)[i];
#pragma unroll
  for (int j = 0; j < 4; ++j) d[j] += fmaxf(v[j] * sc[j] + sh[j], 0.0f);
  ((v4f*)dstbuf)[i] = d;
}

__global__ void final_out_kernel(const float* __restrict__ tmp, const float* __restrict__ wp2,
                                 const float* __restrict__ bp2, float* __restrict__ out, int E) {
  int e = blockIdx.x * 256 + threadIdx.x;
  if (e >= E) return;
  const v4f* row = (const v4f*)(tmp + (size_t)e * 64);
  const v4f* w4 = (const v4f*)wp2;
  float acc = bp2[0];
#pragma unroll
  for (int q = 0; q < 16; ++q) {
    v4f r = row[q], w = w4[q];
    acc += r[0] * w[0] + r[1] * w[1] + r[2] * w[2] + r[3] * w[3];
  }
  out[e] = acc;
}

// ---------------------------------------------------------------------------
// Host orchestration
// ---------------------------------------------------------------------------
extern "C" void kernel_launch(void* const* d_in, const int* in_sizes, int n_in,
                              void* d_out, int out_size, void* d_ws, size_t ws_size,
                              hipStream_t stream) {
  (void)n_in; (void)out_size;
  const float* x_in   = (const float*)d_in[0];
  const float* reads  = (const float*)d_in[2];
  const float* w1n = (const float*)d_in[3];  const float* b1n = (const float*)d_in[4];
  const float* w2n = (const float*)d_in[5];  const float* b2n = (const float*)d_in[6];
  const float* mnw = (const float*)d_in[7];  const float* mwin = (const float*)d_in[8];
  const float* mwconv = (const float*)d_in[9];  const float* mbconv = (const float*)d_in[10];
  const float* mwx = (const float*)d_in[11]; const float* mwdt = (const float*)d_in[12];
  const float* mbdt = (const float*)d_in[13]; const float* malog = (const float*)d_in[14];
  const float* md = (const float*)d_in[15];  const float* mwout = (const float*)d_in[16];
  const float* wbase = (const float*)d_in[17]; const float* bbase = (const float*)d_in[18];
  const float* wm1 = (const float*)d_in[19]; const float* bm1 = (const float*)d_in[20];
  const float* wm2 = (const float*)d_in[21]; const float* bm2 = (const float*)d_in[22];
  const float* we1 = (const float*)d_in[23]; const float* be1 = (const float*)d_in[24];
  const float* we2 = (const float*)d_in[25]; const float* be2 = (const float*)d_in[26];
  const float* a1w = (const float*)d_in[27]; const float* a2w = (const float*)d_in[28];
  const float* a3w = (const float*)d_in[29]; const float* b1w = (const float*)d_in[30];
  const float* b2w = (const float*)d_in[31]; const float* b3w = (const float*)d_in[32];
  const float* a1b = (const float*)d_in[33]; const float* a2b = (const float*)d_in[34];
  const float* a3b = (const float*)d_in[35]; const float* b1b = (const float*)d_in[36];
  const float* b2b = (const float*)d_in[37]; const float* b3b = (const float*)d_in[38];
  const float* bnhb = (const float*)d_in[39]; const float* bneb = (const float*)d_in[40];
  const float* bnhg = (const float*)d_in[41]; const float* bneg = (const float*)d_in[42];
  const float* wp1 = (const float*)d_in[43]; const float* bp1 = (const float*)d_in[44];
  const float* wp2 = (const float*)d_in[45]; const float* bp2 = (const float*)d_in[46];
  const int* read_length = (const int*)d_in[47];
  const int* src = (const int*)d_in[48];
  const int* dst = (const int*)d_in[49];
  float* out = (float*)d_out;

  const int N  = in_sizes[47];
  const int E  = in_sizes[48];
  const int L  = in_sizes[2] / (N * 4);
  const int ML = in_sizes[7] / 4;
  const int NL = in_sizes[33] / 64;

  // ---- workspace layout (floats) ----
  float* ws = (float*)d_ws;
  size_t off = 0;
  auto alloc = [&](size_t n) { size_t o = off; off += n; return o; };
  const size_t o_seq  = alloc((size_t)N * L * 4);
  const size_t o_h    = alloc((size_t)N * 64);
  const size_t o_x2   = alloc((size_t)N * 64);
  const size_t o_hc   = alloc((size_t)N * 64);
  const size_t o_tmpn = alloc((size_t)N * 64);
  const size_t o_P1   = alloc((size_t)N * 64);
  const size_t o_P2   = alloc((size_t)N * 64);
  const size_t o_P3   = alloc((size_t)N * 64);
  const size_t o_Q2   = alloc((size_t)N * 64);
  const size_t o_Q3   = alloc((size_t)N * 64);
  const size_t o_nf   = alloc((size_t)4 * N * 64);  // nf,df,nb,db contiguous
  const size_t o_hagg = alloc((size_t)N * 64);
  const size_t o_cat  = alloc((size_t)N * 128);
  const size_t o_stat = alloc(512);                 // [0..255] edge, [256..511] node
  const size_t o_e    = alloc((size_t)E * 64);
  const size_t o_eh   = alloc((size_t)E * 64);
  // big: data f16 (E*256 halves = E*128 floats) / pdata f16 (E*96 floats) /
  //      mamba temps (9*N*L*8 floats) — take the max
  size_t big_need = (size_t)E * 128;
  if ((size_t)9 * N * L * 8 > big_need) big_need = (size_t)9 * N * L * 8;
  const size_t o_big  = alloc(big_need);
  if (ws_size < off * sizeof(float)) return;  // insufficient scratch (deterministic no-op)

  float* seq  = ws + o_seq;
  float* h    = ws + o_h;
  float* x2   = ws + o_x2;
  float* hc   = ws + o_hc;
  float* tmpn = ws + o_tmpn;
  float* P1 = ws + o_P1; float* P2 = ws + o_P2; float* P3 = ws + o_P3;
  float* Q2 = ws + o_Q2; float* Q3 = ws + o_Q3;
  float* nf = ws + o_nf; float* df = nf + (size_t)N * 64;
  float* nb = df + (size_t)N * 64; float* db = nb + (size_t)N * 64;
  float* hagg = ws + o_hagg;
  float* cat  = ws + o_cat;
  float* stat_e = ws + o_stat;
  float* stat_h = stat_e + 256;
  float* ebuf = ws + o_e;
  float* eh   = ws + o_eh;
  float* big  = ws + o_big;
  _Float16* bigh = (_Float16*)big;   // f16 view for data / pdata

  // Mamba temporaries overlaid inside `big` (used strictly before edge phase)
  const size_t NL8 = (size_t)N * L * 8, NL16 = (size_t)N * L * 16;
  float* xi_pre = big;
  float* zb     = big + NL8;
  float* xic    = big + 2 * NL8;
  float* delt   = big + 3 * NL8;
  float* Bm     = big + 4 * NL8;
  float* Cm     = big + 4 * NL8 + NL16;
  float* ytmp   = big + 4 * NL8 + 2 * NL16;

  const int totNL = N * L;

  // ---- node encoder ----
  node_mlp_kernel<<<N, 128, 0, stream>>>(x_in, w1n, b1n, w2n, b2n, h, N);

  // ---- Mamba encoder over reads ----
  hipMemcpyAsync(seq, reads, (size_t)N * L * 4 * sizeof(float),
                 hipMemcpyDeviceToDevice, stream);
  for (int l = 0; l < ML; ++l) {
    mamba_pre_kernel<<<cdiv(totNL, 256), 256, 0, stream>>>(
        seq, mnw + l * 4, mwin + l * 64, xi_pre, zb, totNL);
    mamba_conv_kernel<<<cdiv(totNL, 256), 256, 0, stream>>>(
        xi_pre, mwconv + l * 32, mbconv + l * 8, mwx + l * 264,
        mwdt + l * 8, mbdt + l * 8, xic, delt, Bm, Cm, L, totNL);
    mamba_scan_kernel<<<cdiv(N * 8, 256), 256, 0, stream>>>(
        xic, delt, Bm, Cm, malog + l * 128, md + l * 8, ytmp, L, N * 8);
    mamba_out_kernel<<<cdiv(totNL, 256), 256, 0, stream>>>(
        ytmp, zb, mwout + l * 32, seq, totNL);
  }
  last_token_kernel<<<cdiv(N * 64, 256), 256, 0, stream>>>(
      seq, read_length, wbase, bbase, x2, N, L);

  // ---- hc = relu(concat(h,x2)@wm1+bm1)@wm2+bm2 ----
  concat2_kernel<<<cdiv(N * 32, 256), 256, 0, stream>>>(h, x2, cat, N);
  gemm64_wmma_kernel<0><<<cdiv(N, 64), 128, 0, stream>>>(cat, wm1, bm1, tmpn, N, 128, 1);
  gemm64_wmma_kernel<0><<<cdiv(N, 64), 128, 0, stream>>>(tmpn, wm2, bm2, hc, N, 64, 0);

  // ---- edge MLP: e = relu(relu(data@we1+be1)@we2+be2), data staged as f16 ----
  gather_edge_kernel<<<cdiv(E * 64, 256), 256, 0, stream>>>(
      h, x2, src, dst, bigh, (long long)E * 64);
  gemm64_wmma_kernel<1><<<cdiv(E, 64), 128, 0, stream>>>(bigh, we1, be1, eh, E, 256, 1);
  gemm64_wmma_kernel<0><<<cdiv(E, 64), 128, 0, stream>>>(eh, we2, be2, ebuf, E, 64, 1);

  // ---- GNN layers ----
  for (int l = 0; l < NL; ++l) {
    zero_kernel<<<cdiv(N * 64, 256), 256, 0, stream>>>(nf, (long long)N * 64);  // 4*N*64/4
    zero_kernel<<<1, 128, 0, stream>>>(stat_e, 128);                            // 512/4
    // edge and node projections
    gemm64_wmma_kernel<0><<<cdiv(E, 64), 128, 0, stream>>>(ebuf, b1w + l * 4096, b1b + l * 64, eh, E, 64, 0);
    gemm64_wmma_kernel<0><<<cdiv(N, 64), 128, 0, stream>>>(hc, a1w + l * 4096, a1b + l * 64, P1, N, 64, 0);
    gemm64_wmma_kernel<0><<<cdiv(N, 64), 128, 0, stream>>>(hc, a2w + l * 4096, a2b + l * 64, P2, N, 64, 0);
    gemm64_wmma_kernel<0><<<cdiv(N, 64), 128, 0, stream>>>(hc, a3w + l * 4096, a3b + l * 64, P3, N, 64, 0);
    gemm64_wmma_kernel<0><<<cdiv(N, 64), 128, 0, stream>>>(hc, b2w + l * 4096, b2b + l * 64, Q2, N, 64, 0);
    gemm64_wmma_kernel<0><<<cdiv(N, 64), 128, 0, stream>>>(hc, b3w + l * 4096, b3b + l * 64, Q3, N, 64, 0);
    // gated message passing + BN stats over edges
    gnn_edge_kernel<<<cdiv(E, 4), 256, 0, stream>>>(
        eh, Q2, Q3, P2, P3, src, dst, nf, df, nb, db, stat_e, E);
    bn_finalize_kernel<<<1, 64, 0, stream>>>(stat_e, bneg + l * 64, bneb + l * 64, 1.0f / (float)E);
    // node aggregation + BN stats over nodes
    node_hagg_kernel<<<cdiv(N * 64, 256), 256, 0, stream>>>(
        P1, nf, df, nb, db, hagg, stat_h, N * 64);
    bn_finalize_kernel<<<1, 64, 0, stream>>>(stat_h, bnhg + l * 64, bnhb + l * 64, 1.0f / (float)N);
    // residual updates
    bn_residual_kernel<<<cdiv(N * 16, 256), 256, 0, stream>>>(hc, hagg, stat_h, (long long)N * 16);
    bn_residual_kernel<<<cdiv(E * 16, 256), 256, 0, stream>>>(ebuf, eh, stat_e, (long long)E * 16);
  }

  // ---- predictor: relu(pdata@wp1+bp1)@wp2+bp2, pdata staged as f16 ----
  gather_pdata_kernel<<<cdiv(E * 48, 256), 256, 0, stream>>>(
      hc, ebuf, src, dst, bigh, (long long)E * 48);
  gemm64_wmma_kernel<1><<<cdiv(E, 64), 128, 0, stream>>>(bigh, wp1, bp1, eh, E, 192, 1);
  final_out_kernel<<<cdiv(E, 256), 256, 0, stream>>>(eh, wp2, bp2, out, E);
}
